// LIIF_fpn_pred_53635551593072
// MI455X (gfx1250) — compile-verified, matrix operand order checked by reading, not run on hardware
//
#include <hip/hip_runtime.h>
#include <hip/hip_bf16.h>

// ---------------------------------------------------------------------------
// LIIF-FPN prediction on gfx1250 (MI455X): bf16 WMMA implicit-GEMM convs +
// WMMA MLP. wave32, v_wmma_f32_16x16x32_bf16, 32x32 register tile per wave,
// double-buffered LDS A-staging, async-to-LDS weight tile for the MLP GEMMs.
// ---------------------------------------------------------------------------

typedef __attribute__((ext_vector_type(8)))  float  v8f;
typedef __attribute__((ext_vector_type(8)))  __bf16 v8bf;
typedef __attribute__((ext_vector_type(16))) __bf16 v16bf;

#define BB    4
#define CC    64
#define HH    48
#define WW    48
#define QQ    16384
#define NQ    (BB * QQ)      // 65536 query rows
#define HID   256
#define KPAD0 288            // 260 padded to multiple of 32
#define WROW  296            // Kpad(max 288) + 8 pad -> conflict-free b128 rows

// ---- fragment load from LDS: 16-bit A/B layout (CDNA5 ISA) ------
// lane%16 = row, lane/16 = khalf: K = {khalf*8..+7} in VGPR0-3, {16+khalf*8..}
// in VGPR4-7 -> two aligned ds_load_b128 per fragment.
__device__ __forceinline__ v16bf lds_frag16(const __bf16* rowp, int khalf) {
  union { v16bf v; v8bf h[2]; } u;
  u.h[0] = *(const v8bf*)(rowp + khalf * 8);
  u.h[1] = *(const v8bf*)(rowp + 16 + khalf * 8);
  return u.v;
}

#define WMMA_BF16(A, B, C) \
  __builtin_amdgcn_wmma_f32_16x16x32_bf16(false, (A), false, (B), (short)0, (C), false, false)

// ---------------------------------------------------------------------------
// Weight packing
// ---------------------------------------------------------------------------
__global__ void pack_conv_w_kernel(const float* __restrict__ w, __bf16* __restrict__ wpk,
                                   int Cout, int Cin, int ks) {
  int idx = blockIdx.x * 256 + threadIdx.x;
  int Ktot = ks * ks * Cin;
  if (idx >= Cout * Ktot) return;
  int co = idx / Ktot, k = idx - co * Ktot;
  int tap = k / Cin, ci = k - tap * Cin;
  int kh = tap / ks, kw = tap - kh * ks;
  wpk[idx] = (__bf16)w[((size_t)(co * Cin + ci) * ks + kh) * ks + kw];
}

__global__ void pack_fc_w_kernel(const float* __restrict__ w, __bf16* __restrict__ wt,
                                 int K, int Cout, int Kpad) {
  int idx = blockIdx.x * 256 + threadIdx.x;
  if (idx >= Cout * Kpad) return;
  int co = idx / Kpad, k = idx - co * Kpad;
  wt[idx] = (__bf16)((k < K) ? w[(size_t)k * Cout + co] : 0.0f);
}

// ---------------------------------------------------------------------------
// Encoder conv 3->64 (tiny K; scalar)
// ---------------------------------------------------------------------------
__global__ void enc_conv_kernel(const float* __restrict__ inp, const float* __restrict__ w,
                                const float* __restrict__ bias, float* __restrict__ out) {
  int idx = blockIdx.x * 256 + threadIdx.x;
  const int tot = BB * CC * HH * WW;
  if (idx >= tot) return;
  int x = idx % WW, t = idx / WW;
  int y = t % HH;  t /= HH;
  int co = t % CC; int b = t / CC;
  float acc = bias[co];
  #pragma unroll
  for (int ci = 0; ci < 3; ++ci)
    #pragma unroll
    for (int kh = 0; kh < 3; ++kh) {
      int iy = y - 1 + kh;
      if ((unsigned)iy >= (unsigned)HH) continue;
      #pragma unroll
      for (int kw = 0; kw < 3; ++kw) {
        int ix = x - 1 + kw;
        if ((unsigned)ix >= (unsigned)WW) continue;
        acc += inp[((size_t)(b * 3 + ci) * HH + iy) * WW + ix] *
               w[((size_t)(co * 3 + ci) * 3 + kh) * 3 + kw];
      }
    }
  out[idx] = acc;
}

// ---------------------------------------------------------------------------
// Implicit-GEMM conv: Out[Cout, Npos] = Wpk[Cout, ks*ks*Cin] x Im2col.
// 256 thr = 8 waves as 2(M) x 4(N); block tile M=64 couts x N=128 positions;
// each wave owns a 32x32 macro-tile = 2x2 WMMA fragments (4 wmma / K-chunk).
// Double-buffered LDS, register staging, one barrier per chunk.
// mode 0 = plain NCHW store, 1 = fused pixel-shuffle(r=2), 2 = accumulate.
// ---------------------------------------------------------------------------
__global__ __launch_bounds__(256)
void conv_wmma_kernel(const float* __restrict__ src, const __bf16* __restrict__ wpk,
                      const float* __restrict__ bias, float* __restrict__ dst,
                      int Bn, int Cin, int Hin, int Win,
                      int Cout, int Hout, int Wout,
                      int ks, int stride, int pad, int mode) {
  const int tid  = threadIdx.x;
  const int lane = tid & 31;
  const int wv   = tid >> 5;
  const int m0   = (wv & 1) * 32;        // 2 wave-rows cover 64 couts
  const int n0   = (wv >> 1) * 32;       // 4 wave-cols cover 128 positions
  const int Ktot = ks * ks * Cin;
  const int Npos = Bn * Hout * Wout;
  const int nBase = blockIdx.x * 128;
  const int coutBase = blockIdx.y * 64;
  const int HWin = Hin * Win;
  const int HWout = Hout * Wout;

  __shared__ __bf16 Asm[2][64][40];      // weights tile (padded rows, 80B)
  __shared__ __bf16 Bsm[2][128][40];     // im2col tile

  // ---- staging assignments -----------------------------------------------
  // A: 64x32 = 2048 bf16 / 256 thr -> one v8bf each
  const int arow  = tid >> 2;
  const int akoff = (tid & 3) * 8;
  const __bf16* aptr = wpk + (size_t)(coutBase + arow) * Ktot + akoff;
  // B: 128x32 = 4096 / 256 thr -> 16 consecutive input channels, 1 position
  const int nl   = tid >> 1;
  const int koff = (tid & 1) * 16;
  int nb = nBase + nl;
  bool nvalid = nb < Npos;
  int bb = 0, oy = 0, ox = 0;
  if (nvalid) { bb = nb / HWout; int r = nb - bb * HWout; oy = r / Wout; ox = r - oy * Wout; }

  v8bf  areg = {};
  float breg[16];

  auto stageB = [&](int kb) {
    int kg  = kb + koff;                 // 16 consecutive k, same 3x3 tap
    int tap = kg / Cin;
    int cib = kg - tap * Cin;
    #pragma unroll
    for (int j = 0; j < 16; ++j) breg[j] = 0.0f;
    if (nvalid) {
      int kh = tap / ks, kw = tap - kh * ks;
      int iy = oy * stride - pad + kh;
      int ix = ox * stride - pad + kw;
      if ((unsigned)iy < (unsigned)Hin && (unsigned)ix < (unsigned)Win) {
        const float* p = src + ((size_t)(bb * Cin + cib) * Hin + iy) * Win + ix;
        #pragma unroll
        for (int j = 0; j < 16; ++j) breg[j] = p[(size_t)j * HWin];
      }
    }
  };

  // ---- prologue -----------------------------------------------------------
  areg = *(const v8bf*)(aptr + 0);
  stageB(0);

  v8f acc[2][2] = {};
  const int rfrag = lane & 15, khalf = lane >> 4;
  int buf = 0;

  for (int kb = 0; kb < Ktot; kb += 32) {
    *(v8bf*)&Asm[buf][arow][akoff] = areg;
    #pragma unroll
    for (int j = 0; j < 16; ++j) Bsm[buf][nl][koff + j] = (__bf16)breg[j];
    __syncthreads();
    if (kb + 32 < Ktot) {                 // overlap next chunk's global loads
      areg = *(const v8bf*)(aptr + kb + 32);
      __builtin_prefetch(aptr + kb + 64, 0, 0);
      stageB(kb + 32);
    }
    v16bf a0 = lds_frag16(&Asm[buf][m0 + rfrag][0], khalf);
    v16bf a1 = lds_frag16(&Asm[buf][m0 + 16 + rfrag][0], khalf);
    v16bf b0 = lds_frag16(&Bsm[buf][n0 + rfrag][0], khalf);
    v16bf b1 = lds_frag16(&Bsm[buf][n0 + 16 + rfrag][0], khalf);
    acc[0][0] = WMMA_BF16(a0, b0, acc[0][0]);
    acc[0][1] = WMMA_BF16(a0, b1, acc[0][1]);
    acc[1][0] = WMMA_BF16(a1, b0, acc[1][0]);
    acc[1][1] = WMMA_BF16(a1, b1, acc[1][1]);
    buf ^= 1;
  }

  // ---- store: lane = col N (lane%16), VGPR v = row M = v + 8*(lane/16) ----
  #pragma unroll
  for (int mi = 0; mi < 2; ++mi)
    #pragma unroll
    for (int ni = 0; ni < 2; ++ni)
      #pragma unroll
      for (int v = 0; v < 8; ++v) {
        int gm = coutBase + m0 + mi * 16 + v + 8 * khalf;
        int gn = nBase + n0 + ni * 16 + rfrag;
        if (gn < Npos) {
          int b2 = gn / HWout;
          int rr = gn - b2 * HWout;
          int y = rr / Wout, x = rr - y * Wout;
          float val = acc[mi][ni][v] + bias[gm];
          if (mode == 1) {  // fused pixel-shuffle r=2
            int cc = gm >> 2, ri = (gm >> 1) & 1, rj = gm & 1;
            dst[(((size_t)b2 * (Cout >> 2) + cc) * (Hout * 2) + (y * 2 + ri)) * (Wout * 2) +
                (x * 2 + rj)] = val;
          } else {
            size_t o = (((size_t)b2 * Cout + gm) * Hout + y) * Wout + x;
            if (mode == 2) val += dst[o];
            dst[o] = val;
          }
        }
      }
}

// ---------------------------------------------------------------------------
// FC WMMA: Y[N, Cout] = act(X[N, Kpad] * Wt[Cout, Kpad]^T + b), bf16 out.
// 8 waves as 4(M) x 2(N); block tile M=128 rows x N=64 couts; 32x32 per wave.
// The whole 64 x Kpad weight tile is copied LDS-resident ONCE per block via
// per-lane global_load_async_to_lds_b128 (ASYNCcnt), fenced with
// s_wait_asynccnt + barrier; only activations are double-buffered per chunk.
// ---------------------------------------------------------------------------
__global__ __launch_bounds__(256)
void fc_wmma_kernel(const __bf16* __restrict__ X, int Nrows, int Kpad,
                    const __bf16* __restrict__ Wt, const float* __restrict__ bias,
                    int Cout, __bf16* __restrict__ Y, int relu) {
  const int tid  = threadIdx.x;
  const int lane = tid & 31;
  const int wv   = tid >> 5;
  const int m0   = (wv & 3) * 32;        // 4 wave-rows cover 128 rows
  const int n0   = (wv >> 2) * 32;       // 2 wave-cols cover 64 couts
  const int rowBase  = blockIdx.x * 128;
  const int coutBase = blockIdx.y * 64;

  __shared__ __bf16 Asm[2][128][40];     // activation tiles (double-buffered)
  __shared__ __bf16 Wsm[64][WROW];       // whole weight tile, async-resident

  // ---- async copy of the 64 x Kpad weight tile into LDS -------------------
  {
    const int chunks = Kpad >> 3;        // 16B units per weight row
    unsigned wsmBase = (unsigned)(size_t)(&Wsm[0][0]);
    for (int idx = tid; idx < 64 * chunks; idx += 256) {
      int n = idx / chunks, j = idx - n * chunks;
      unsigned ldsa = wsmBase + (unsigned)(n * WROW + j * 8) * 2u;
      unsigned goff = (unsigned)(((coutBase + n) * Kpad + j * 8) * 2);
      asm volatile("global_load_async_to_lds_b128 %0, %1, %2"
                   :: "v"(ldsa), "v"(goff), "s"(Wt)
                   : "memory");
    }
  }

  // A: 128x32 = 4096 / 256 thr -> 16 contiguous bf16 (two v8bf)
  const int arow  = tid >> 1;
  const int akoff = (tid & 1) * 16;
  const __bf16* aptr = X + (size_t)(rowBase + arow) * Kpad + akoff;

  v8bf areg0, areg1;
  areg0 = *(const v8bf*)(aptr + 0);
  areg1 = *(const v8bf*)(aptr + 8);

  // own-wave async ops done; the barrier inside the loop publishes all waves'
  asm volatile("s_wait_asynccnt 0x0" ::: "memory");

  v8f acc[2][2] = {};
  const int rfrag = lane & 15, khalf = lane >> 4;
  int buf = 0;

  for (int kb = 0; kb < Kpad; kb += 32) {
    *(v8bf*)&Asm[buf][arow][akoff]     = areg0;
    *(v8bf*)&Asm[buf][arow][akoff + 8] = areg1;
    __syncthreads();
    if (kb + 32 < Kpad) {
      areg0 = *(const v8bf*)(aptr + kb + 32);
      areg1 = *(const v8bf*)(aptr + kb + 40);
      __builtin_prefetch(aptr + kb + 64, 0, 0);
    }
    v16bf a0 = lds_frag16(&Asm[buf][m0 + rfrag][0], khalf);
    v16bf a1 = lds_frag16(&Asm[buf][m0 + 16 + rfrag][0], khalf);
    v16bf b0 = lds_frag16(&Wsm[n0 + rfrag][kb], khalf);
    v16bf b1 = lds_frag16(&Wsm[n0 + 16 + rfrag][kb], khalf);
    acc[0][0] = WMMA_BF16(a0, b0, acc[0][0]);
    acc[0][1] = WMMA_BF16(a0, b1, acc[0][1]);
    acc[1][0] = WMMA_BF16(a1, b0, acc[1][0]);
    acc[1][1] = WMMA_BF16(a1, b1, acc[1][1]);
    buf ^= 1;
  }

  #pragma unroll
  for (int mi = 0; mi < 2; ++mi)
    #pragma unroll
    for (int ni = 0; ni < 2; ++ni)
      #pragma unroll
      for (int v = 0; v < 8; ++v) {
        int row  = rowBase + m0 + mi * 16 + v + 8 * khalf;
        int cout = coutBase + n0 + ni * 16 + rfrag;
        if (row < Nrows) {
          float val = acc[mi][ni][v] + bias[cout];
          if (relu) val = fmaxf(val, 0.0f);
          Y[(size_t)row * Cout + cout] = (__bf16)val;
        }
      }
}

// ---------------------------------------------------------------------------
// nearest-grid-sample index (align_corners=False, border clamp, nearbyint)
// ---------------------------------------------------------------------------
__device__ __forceinline__ int nearest_idx(float c, int n) {
  float f = rintf((c + 1.0f) * (0.5f * (float)n) - 0.5f);
  f = fminf(fmaxf(f, 0.0f), (float)(n - 1));
  return (int)f;
}

// ---------------------------------------------------------------------------
// Build MLP input rows for one ensemble shift; also emits area weights.
// X row: [feat(64) | feat5(64) | feat4(64) | feat3(64) | rel(2) | relcell(2) | pad]
// ---------------------------------------------------------------------------
__global__ void build_x_kernel(const float* __restrict__ feat, const float* __restrict__ feat3,
                               const float* __restrict__ feat4, const float* __restrict__ feat5,
                               const float* __restrict__ coord, const float* __restrict__ cell,
                               __bf16* __restrict__ X, float* __restrict__ areas, int s) {
  int t = blockIdx.x * 256 + threadIdx.x;
  if (t >= NQ) return;
  int b = t / QQ;
  float cy = coord[(size_t)t * 2 + 0];
  float cx = coord[(size_t)t * 2 + 1];
  float vx = (s < 2) ? -1.0f : 1.0f;
  float vy = (s & 1) ? 1.0f : -1.0f;
  const float rx = 1.0f / (float)HH, ry = 1.0f / (float)WW, eps = 1e-6f;
  float sy = fminf(fmaxf(cy + vx * rx + eps, -1.0f + 1e-6f), 1.0f - 1e-6f);
  float sx = fminf(fmaxf(cx + vy * ry + eps, -1.0f + 1e-6f), 1.0f - 1e-6f);

  __bf16* xr = X + (size_t)t * KPAD0;

  int iy = nearest_idx(sy, HH), ix = nearest_idx(sx, WW);
  const float* p0 = feat + ((size_t)b * CC * HH * WW) + (size_t)iy * WW + ix;
  for (int c = 0; c < CC; ++c) xr[c] = (__bf16)p0[(size_t)c * HH * WW];

  int i5y = nearest_idx(sy, 2 * HH), i5x = nearest_idx(sx, 2 * WW);
  const float* p5 = feat5 + ((size_t)b * CC * 4 * HH * WW) + (size_t)i5y * (2 * WW) + i5x;
  for (int c = 0; c < CC; ++c) xr[64 + c] = (__bf16)p5[(size_t)c * 4 * HH * WW];

  int i4y = nearest_idx(sy, 4 * HH), i4x = nearest_idx(sx, 4 * WW);
  const float* p4 = feat4 + ((size_t)b * CC * 16 * HH * WW) + (size_t)i4y * (4 * WW) + i4x;
  for (int c = 0; c < CC; ++c) xr[128 + c] = (__bf16)p4[(size_t)c * 16 * HH * WW];

  int i3y = nearest_idx(sy, 8 * HH), i3x = nearest_idx(sx, 8 * WW);
  const float* p3 = feat3 + ((size_t)b * CC * 64 * HH * WW) + (size_t)i3y * (8 * WW) + i3x;
  for (int c = 0; c < CC; ++c) xr[192 + c] = (__bf16)p3[(size_t)c * 64 * HH * WW];

  float qcy = -1.0f + (2.0f * (float)iy + 1.0f) / (float)HH;
  float qcx = -1.0f + (2.0f * (float)ix + 1.0f) / (float)WW;
  float rel0 = (cy - qcy) * (float)HH;
  float rel1 = (cx - qcx) * (float)WW;
  xr[256] = (__bf16)rel0;
  xr[257] = (__bf16)rel1;
  xr[258] = (__bf16)(cell[(size_t)t * 2 + 0] * (float)HH);
  xr[259] = (__bf16)(cell[(size_t)t * 2 + 1] * (float)WW);
  #pragma unroll
  for (int k = 260; k < KPAD0; ++k) xr[k] = (__bf16)0.0f;

  areas[t] = fabsf(rel0 * rel1) + 1e-9f;
}

// ---------------------------------------------------------------------------
// MLP head 256 -> 3 (tiny)
// ---------------------------------------------------------------------------
__global__ void fc_last_kernel(const __bf16* __restrict__ H, const float* __restrict__ w,
                               const float* __restrict__ b, float* __restrict__ pred) {
  int t = blockIdx.x * 256 + threadIdx.x;
  if (t >= NQ) return;
  const __bf16* hr = H + (size_t)t * HID;
  float a0 = b[0], a1 = b[1], a2 = b[2];
  for (int k = 0; k < HID; ++k) {
    float hv = (float)hr[k];
    a0 += hv * w[k * 3 + 0];
    a1 += hv * w[k * 3 + 1];
    a2 += hv * w[k * 3 + 2];
  }
  pred[(size_t)t * 3 + 0] = a0;
  pred[(size_t)t * 3 + 1] = a1;
  pred[(size_t)t * 3 + 2] = a2;
}

// ---------------------------------------------------------------------------
// Local ensemble (diagonal-swapped areas) + bilinear border residual
// ---------------------------------------------------------------------------
__global__ void combine_kernel(const float* __restrict__ coord, const float* __restrict__ areas,
                               const float* __restrict__ preds, const float* __restrict__ inp,
                               float* __restrict__ out) {
  int t = blockIdx.x * 256 + threadIdx.x;
  if (t >= NQ) return;
  int b = t / QQ;
  float a0 = areas[0 * NQ + t], a1 = areas[1 * NQ + t];
  float a2 = areas[2 * NQ + t], a3 = areas[3 * NQ + t];
  float tot = a0 + a1 + a2 + a3;
  float w0 = a3 / tot, w1 = a2 / tot, w2 = a1 / tot, w3 = a0 / tot;

  float cy = coord[(size_t)t * 2 + 0];
  float cx = coord[(size_t)t * 2 + 1];
  float fy = fminf(fmaxf((cy + 1.0f) * (0.5f * HH) - 0.5f, 0.0f), (float)(HH - 1));
  float fx = fminf(fmaxf((cx + 1.0f) * (0.5f * WW) - 0.5f, 0.0f), (float)(WW - 1));
  float y0f = floorf(fy), x0f = floorf(fx);
  float wy = fy - y0f, wx = fx - x0f;
  int y0 = (int)y0f, x0 = (int)x0f;
  int y1 = min(y0 + 1, HH - 1), x1 = min(x0 + 1, WW - 1);

  #pragma unroll
  for (int c = 0; c < 3; ++c) {
    float r = preds[(size_t)(0 * NQ + t) * 3 + c] * w0 +
              preds[(size_t)(1 * NQ + t) * 3 + c] * w1 +
              preds[(size_t)(2 * NQ + t) * 3 + c] * w2 +
              preds[(size_t)(3 * NQ + t) * 3 + c] * w3;
    const float* ip = inp + (size_t)(b * 3 + c) * HH * WW;
    float v00 = ip[y0 * WW + x0], v01 = ip[y0 * WW + x1];
    float v10 = ip[y1 * WW + x0], v11 = ip[y1 * WW + x1];
    r += v00 * (1.f - wy) * (1.f - wx) + v01 * (1.f - wy) * wx +
         v10 * wy * (1.f - wx) + v11 * wy * wx;
    out[(size_t)t * 3 + c] = r;
  }
}

// ---------------------------------------------------------------------------
// Host driver
// ---------------------------------------------------------------------------
extern "C" void kernel_launch(void* const* d_in, const int* in_sizes, int n_in,
                              void* d_out, int out_size, void* d_ws, size_t ws_size,
                              hipStream_t stream) {
  const float* inp     = (const float*)d_in[0];
  const float* coord   = (const float*)d_in[1];
  const float* cell    = (const float*)d_in[2];
  const float* enc_w   = (const float*)d_in[3];
  const float* enc_b   = (const float*)d_in[4];
  const float* up1_w   = (const float*)d_in[5];
  const float* up1_b   = (const float*)d_in[6];
  const float* up2_w   = (const float*)d_in[7];
  const float* up2_b   = (const float*)d_in[8];
  const float* up3_w   = (const float*)d_in[9];
  const float* up3_b   = (const float*)d_in[10];
  const float* skip1_w = (const float*)d_in[11];
  const float* skip1_b = (const float*)d_in[12];
  const float* skip2_w = (const float*)d_in[13];
  const float* skip2_b = (const float*)d_in[14];
  const float* down4_w = (const float*)d_in[15];
  const float* down4_b = (const float*)d_in[16];
  const float* down5_w = (const float*)d_in[17];
  const float* down5_b = (const float*)d_in[18];
  const float* mlp_w0  = (const float*)d_in[19];
  const float* mlp_b0  = (const float*)d_in[20];
  const float* mlp_w1  = (const float*)d_in[21];
  const float* mlp_b1  = (const float*)d_in[22];
  const float* mlp_w2  = (const float*)d_in[23];
  const float* mlp_b2  = (const float*)d_in[24];
  const float* mlp_w3  = (const float*)d_in[25];
  const float* mlp_b3  = (const float*)d_in[26];
  const float* mlp_w4  = (const float*)d_in[27];
  const float* mlp_b4  = (const float*)d_in[28];

  char* wsb = (char*)d_ws;
  size_t off = 0;
  auto alloc = [&](size_t bytes) -> void* {
    void* p = wsb + off;
    off += (bytes + 255) & ~(size_t)255;
    return p;
  };

  float*  feat  = (float*)alloc((size_t)BB * CC * HH * WW * 4);           // 48^2
  float*  feat1 = (float*)alloc((size_t)BB * CC * 4  * HH * WW * 4);      // 96^2
  float*  feat2 = (float*)alloc((size_t)BB * CC * 16 * HH * WW * 4);      // 192^2
  float*  feat3 = (float*)alloc((size_t)BB * CC * 64 * HH * WW * 4);      // 384^2
  float*  feat4 = (float*)alloc((size_t)BB * CC * 16 * HH * WW * 4);      // 192^2
  float*  feat5 = (float*)alloc((size_t)BB * CC * 4  * HH * WW * 4);      // 96^2
  __bf16* wup1  = (__bf16*)alloc((size_t)256 * 576 * 2);
  __bf16* wup2  = (__bf16*)alloc((size_t)256 * 576 * 2);
  __bf16* wup3  = (__bf16*)alloc((size_t)256 * 576 * 2);
  __bf16* wdn4  = (__bf16*)alloc((size_t)64 * 576 * 2);
  __bf16* wdn5  = (__bf16*)alloc((size_t)64 * 576 * 2);
  __bf16* wsk1  = (__bf16*)alloc((size_t)64 * 64 * 2);
  __bf16* wsk2  = (__bf16*)alloc((size_t)64 * 64 * 2);
  __bf16* wf0   = (__bf16*)alloc((size_t)HID * KPAD0 * 2);
  __bf16* wf1   = (__bf16*)alloc((size_t)HID * HID * 2);
  __bf16* wf2   = (__bf16*)alloc((size_t)HID * HID * 2);
  __bf16* wf3   = (__bf16*)alloc((size_t)HID * HID * 2);
  __bf16* Xb    = (__bf16*)alloc((size_t)NQ * KPAD0 * 2);
  __bf16* Ha    = (__bf16*)alloc((size_t)NQ * HID * 2);
  __bf16* Hb    = (__bf16*)alloc((size_t)NQ * HID * 2);
  float*  preds = (float*)alloc((size_t)4 * NQ * 3 * 4);
  float*  areas = (float*)alloc((size_t)4 * NQ * 4);
  (void)ws_size; (void)n_in; (void)in_sizes; (void)out_size;

  // ---- weight packing -----------------------------------------------------
  pack_conv_w_kernel<<<(256 * 576 + 255) / 256, 256, 0, stream>>>(up1_w, wup1, 256, 64, 3);
  pack_conv_w_kernel<<<(256 * 576 + 255) / 256, 256, 0, stream>>>(up2_w, wup2, 256, 64, 3);
  pack_conv_w_kernel<<<(256 * 576 + 255) / 256, 256, 0, stream>>>(up3_w, wup3, 256, 64, 3);
  pack_conv_w_kernel<<<(64 * 576 + 255) / 256, 256, 0, stream>>>(down4_w, wdn4, 64, 64, 3);
  pack_conv_w_kernel<<<(64 * 576 + 255) / 256, 256, 0, stream>>>(down5_w, wdn5, 64, 64, 3);
  pack_conv_w_kernel<<<(64 * 64 + 255) / 256, 256, 0, stream>>>(skip1_w, wsk1, 64, 64, 1);
  pack_conv_w_kernel<<<(64 * 64 + 255) / 256, 256, 0, stream>>>(skip2_w, wsk2, 64, 64, 1);
  pack_fc_w_kernel<<<(HID * KPAD0 + 255) / 256, 256, 0, stream>>>(mlp_w0, wf0, 260, HID, KPAD0);
  pack_fc_w_kernel<<<(HID * HID + 255) / 256, 256, 0, stream>>>(mlp_w1, wf1, HID, HID, HID);
  pack_fc_w_kernel<<<(HID * HID + 255) / 256, 256, 0, stream>>>(mlp_w2, wf2, HID, HID, HID);
  pack_fc_w_kernel<<<(HID * HID + 255) / 256, 256, 0, stream>>>(mlp_w3, wf3, HID, HID, HID);

  // ---- feature pyramid ----------------------------------------------------
  enc_conv_kernel<<<(BB * CC * HH * WW + 255) / 256, 256, 0, stream>>>(inp, enc_w, enc_b, feat);

  // up1: 64->256 @48^2, pixel-shuffle -> feat1 [4,64,96,96]
  conv_wmma_kernel<<<dim3((BB * HH * WW) / 128, 4), 256, 0, stream>>>(
      feat, wup1, up1_b, feat1, BB, 64, HH, WW, 256, HH, WW, 3, 1, 1, 1);
  // up2 @96^2 -> feat2 [4,64,192,192]
  conv_wmma_kernel<<<dim3((BB * 4 * HH * WW) / 128, 4), 256, 0, stream>>>(
      feat1, wup2, up2_b, feat2, BB, 64, 2 * HH, 2 * WW, 256, 2 * HH, 2 * WW, 3, 1, 1, 1);
  // up3 @192^2 -> feat3 [4,64,384,384]
  conv_wmma_kernel<<<dim3((BB * 16 * HH * WW) / 128, 4), 256, 0, stream>>>(
      feat2, wup3, up3_b, feat3, BB, 64, 4 * HH, 4 * WW, 256, 4 * HH, 4 * WW, 3, 1, 1, 1);
  // skip1 (1x1 on feat2) -> feat4, then down4 (3x3 stride2 on feat3) += feat4
  conv_wmma_kernel<<<dim3((BB * 16 * HH * WW) / 128, 1), 256, 0, stream>>>(
      feat2, wsk1, skip1_b, feat4, BB, 64, 4 * HH, 4 * WW, 64, 4 * HH, 4 * WW, 1, 1, 0, 0);
  conv_wmma_kernel<<<dim3((BB * 16 * HH * WW) / 128, 1), 256, 0, stream>>>(
      feat3, wdn4, down4_b, feat4, BB, 64, 8 * HH, 8 * WW, 64, 4 * HH, 4 * WW, 3, 2, 1, 2);
  // skip2 (1x1 on feat1) -> feat5, then down5 (3x3 stride2 on feat4) += feat5
  conv_wmma_kernel<<<dim3((BB * 4 * HH * WW) / 128, 1), 256, 0, stream>>>(
      feat1, wsk2, skip2_b, feat5, BB, 64, 2 * HH, 2 * WW, 64, 2 * HH, 2 * WW, 1, 1, 0, 0);
  conv_wmma_kernel<<<dim3((BB * 4 * HH * WW) / 128, 1), 256, 0, stream>>>(
      feat4, wdn5, down5_b, feat5, BB, 64, 4 * HH, 4 * WW, 64, 2 * HH, 2 * WW, 3, 2, 1, 2);

  // ---- local-ensemble MLP (4 shifts) --------------------------------------
  for (int s = 0; s < 4; ++s) {
    build_x_kernel<<<(NQ + 255) / 256, 256, 0, stream>>>(feat, feat3, feat4, feat5, coord, cell,
                                                         Xb, areas + (size_t)s * NQ, s);
    fc_wmma_kernel<<<dim3(NQ / 128, HID / 64), 256, 0, stream>>>(Xb, NQ, KPAD0, wf0, mlp_b0,
                                                                 HID, Ha, 1);
    fc_wmma_kernel<<<dim3(NQ / 128, HID / 64), 256, 0, stream>>>(Ha, NQ, HID, wf1, mlp_b1,
                                                                 HID, Hb, 1);
    fc_wmma_kernel<<<dim3(NQ / 128, HID / 64), 256, 0, stream>>>(Hb, NQ, HID, wf2, mlp_b2,
                                                                 HID, Ha, 1);
    fc_wmma_kernel<<<dim3(NQ / 128, HID / 64), 256, 0, stream>>>(Ha, NQ, HID, wf3, mlp_b3,
                                                                 HID, Hb, 1);
    fc_last_kernel<<<(NQ + 255) / 256, 256, 0, stream>>>(Hb, mlp_w4, mlp_b4,
                                                         preds + (size_t)s * NQ * 3);
  }

  combine_kernel<<<(NQ + 255) / 256, 256, 0, stream>>>(coord, areas, preds, inp,
                                                       (float*)d_out);
}
